// UpLayer_56341380989521
// MI455X (gfx1250) — matrix operand order repaired and checked by reading, not compiled, not run on hardware
//
#include <hip/hip_runtime.h>

typedef __attribute__((ext_vector_type(16))) _Float16 v16h;
typedef __attribute__((ext_vector_type(8)))  _Float16 v8h;
typedef __attribute__((ext_vector_type(4)))  _Float16 v4h;
typedef __attribute__((ext_vector_type(8)))  float    v8f;

#define HIDDEN    4096
#define REMAINED  4403
#define BATCH     64
#define KC        64      // K chunk staged in LDS per iteration
#define NT        16      // N columns per workgroup (one WMMA tile wide)
#define LDSW      72      // padded LDS row stride in halves (144B = 36 banks -> conflict-free TR reads)

// 128 threads = 4 waves. Wave w owns the 16x16 C tile for M rows [16w,16w+16).
__global__ __launch_bounds__(128)
void gather_gemm_wmma(const float* __restrict__ x,
                      const float* __restrict__ w,
                      const int*   __restrict__ idx,
                      float* __restrict__ out0,
                      float* __restrict__ out1,
                      int kper)
{
    __shared__ _Float16 Ash[BATCH * LDSW];   // X chunk   [M=64][KC] as f16
    __shared__ _Float16 Bsh[NT * LDSW];      // gathered W chunk [N=16][KC] as f16

    const int tid  = threadIdx.x;
    const int wave = tid >> 5;
    const int lane = tid & 31;
    const int half = lane >> 4;
    const int l16  = lane & 15;

    const int nbase  = blockIdx.x * NT;
    const int kbase  = blockIdx.y * kper;
    const int nchunk = kper / KC;

    // ---- cooperative global->LDS staging assignment ----
    const int f4   = tid & 15;   // which 16B (float4) of the 64-float row chunk
    const int rowg = tid >> 4;   // 0..7

    const float* aptr[8];
#pragma unroll
    for (int p = 0; p < 8; ++p)
        aptr[p] = x + (size_t)(rowg + 8 * p) * HIDDEN + kbase + f4 * 4;

    const float* bptr[2];
#pragma unroll
    for (int p = 0; p < 2; ++p) {
        int n = nbase + rowg + 8 * p;
        n = (n < REMAINED) ? n : (REMAINED - 1);   // clamp pad columns
        bptr[p] = w + (size_t)idx[n] * HIDDEN + kbase + f4 * 4;
    }

    // prefetch chunk 0 (global_load_b128, fully coalesced along each row)
    float4 areg[8], breg[2];
#pragma unroll
    for (int p = 0; p < 8; ++p) areg[p] = *(const float4*)aptr[p];
#pragma unroll
    for (int p = 0; p < 2; ++p) breg[p] = *(const float4*)bptr[p];

    v8f acc = {};

    // per-lane LDS fragment base addresses (16-bit WMMA A/B layouts, ISA 7.12.2)
    const _Float16* afrag = Ash + (wave * 16 + l16) * LDSW + half * 8;  // A: K runs {8h..8h+7},{16+8h..}
    const _Float16* bfrag = Bsh + l16 * LDSW + half * 8;                // B: K runs {8h..8h+7},{16+8h..}

    for (int c = 0; c < nchunk; ++c) {
        __syncthreads();   // previous chunk's compute done before overwrite
#pragma unroll
        for (int p = 0; p < 8; ++p) {
            float4 v = areg[p];
            v4h h = { (_Float16)v.x, (_Float16)v.y, (_Float16)v.z, (_Float16)v.w };
            *(v4h*)(Ash + (rowg + 8 * p) * LDSW + f4 * 4) = h;
        }
#pragma unroll
        for (int p = 0; p < 2; ++p) {
            float4 v = breg[p];
            v4h h = { (_Float16)v.x, (_Float16)v.y, (_Float16)v.z, (_Float16)v.w };
            *(v4h*)(Bsh + (rowg + 8 * p) * LDSW + f4 * 4) = h;
        }
        __syncthreads();

        // software pipeline: issue next chunk's global loads before computing
        if (c + 1 < nchunk) {
            const int off = (c + 1) * KC;
#pragma unroll
            for (int p = 0; p < 8; ++p) areg[p] = *(const float4*)(aptr[p] + off);
#pragma unroll
            for (int p = 0; p < 2; ++p) breg[p] = *(const float4*)(bptr[p] + off);
        }

        // two 16x16x32 WMMA K-steps per 64-wide chunk
#pragma unroll
        for (int ks = 0; ks < 2; ++ks) {
            const _Float16* ab = afrag + ks * 32;
            v8h alo = *(const v8h*)(ab);          // ds_load_b128
            v8h ahi = *(const v8h*)(ab + 16);     // ds_load_b128
            const _Float16* bb = bfrag + ks * 32;
            v8h blo = *(const v8h*)(bb);
            v8h bhi = *(const v8h*)(bb + 16);
            v16h A = __builtin_shufflevector(alo, ahi, 0,1,2,3,4,5,6,7,8,9,10,11,12,13,14,15);
            v16h B = __builtin_shufflevector(blo, bhi, 0,1,2,3,4,5,6,7,8,9,10,11,12,13,14,15);
            acc = __builtin_amdgcn_wmma_f32_16x16x32_f16(
                      /*neg_a=*/false, A, /*neg_b=*/false, B,
                      /*c_mod=*/(short)0, acc, /*reuse_a=*/false, /*reuse_b=*/false);
        }
    }

    // epilogue: C layout = VGPR r -> M=r (lanes 0-15) / M=r+8 (lanes 16-31), N=l16
    float* outp = (blockIdx.y == 0) ? out0 : out1;
    const int ncol = nbase + l16;
    if (ncol < REMAINED) {
#pragma unroll
        for (int r = 0; r < 8; ++r) {
            const int m = wave * 16 + 8 * half + r;
            outp[(size_t)m * REMAINED + ncol] = acc[r];
        }
    }
}

__global__ void reduce_add(float* __restrict__ out, const float* __restrict__ part, int n)
{
    int i = blockIdx.x * blockDim.x + threadIdx.x;
    if (i < n) out[i] += part[i];
}

extern "C" void kernel_launch(void* const* d_in, const int* in_sizes, int n_in,
                              void* d_out, int out_size, void* d_ws, size_t ws_size,
                              hipStream_t stream)
{
    const float* x   = (const float*)d_in[0];
    const float* wgt = (const float*)d_in[1];
    const int*   idx = (const int*)d_in[2];
    float* out  = (float*)d_out;
    float* part = (float*)d_ws;

    const int ntiles = (REMAINED + NT - 1) / NT;          // 276
    const size_t need = (size_t)BATCH * REMAINED * sizeof(float);
    const int ksplit = (ws_size >= need && part != nullptr) ? 2 : 1;
    const int kper = HIDDEN / ksplit;

    dim3 grid(ntiles, ksplit);
    gather_gemm_wmma<<<grid, 128, 0, stream>>>(x, wgt, idx, out, part, kper);

    if (ksplit == 2) {
        const int n = BATCH * REMAINED;
        reduce_add<<<(n + 255) / 256, 256, 0, stream>>>(out, part, n);
    }
}
